// Cp_linear_2D_6760278524110
// MI455X (gfx1250) — matrix operand order repaired and checked by reading, not compiled, not run on hardware
//
#include <hip/hip_runtime.h>

// CP (Kruskal) linear layer, factorized:
//   stage 0: U1t = (U1 * lam)^T              [R x IN]   (transpose + scale fuse)
//   stage 1: T   = x @ U1t^T                 [B x R]
//   stage 2: out = T @ U2^T + bias           [B x OUT]
// GEMM stages: bf16 WMMA (f32 accumulate), double-buffered K-tiles staged by the
// Tensor Data Mover (tensor_load_to_lds + s_wait_tensorcnt) on gfx1250 wave32,
// with async global->LDS b128 loads as fallback when the TDM builtin is absent.

typedef __attribute__((ext_vector_type(16))) __bf16 v16bf;
typedef __attribute__((ext_vector_type(8)))  float  v8f;
typedef unsigned int u32x4 __attribute__((ext_vector_type(4)));
typedef int          i32x4 __attribute__((ext_vector_type(4)));
typedef int          i32x8 __attribute__((ext_vector_type(8)));

#define BM 128
#define BN 64
#define BK 32
#define PAD 4          // matches TDM pad_amount = 4 DWORDs every 32 DWORDs
#define THREADS 128    // 4 waves of 32

#if __has_builtin(__builtin_amdgcn_tensor_load_to_lds)
#define USE_TDM 1
#else
#define USE_TDM 0
#endif

__device__ __forceinline__ __bf16 f2bf(float f) { return (__bf16)f; }

__device__ __forceinline__ void pack8(v16bf& v, int base, const float4& p, const float4& q) {
  v[base + 0] = f2bf(p.x); v[base + 1] = f2bf(p.y);
  v[base + 2] = f2bf(p.z); v[base + 3] = f2bf(p.w);
  v[base + 4] = f2bf(q.x); v[base + 5] = f2bf(q.y);
  v[base + 6] = f2bf(q.z); v[base + 7] = f2bf(q.w);
}

#if USE_TDM
// One TDM descriptor: 2D tile (tile_d0 x tile_d1) of f32, row stride `stride_elems`,
// into LDS at lds_off with +4 DWORD padding after every 32 DWORDs.
__device__ __forceinline__ void tdm_load_2d(unsigned lds_off, const float* gaddr,
                                            unsigned tile_d0, unsigned tile_d1,
                                            unsigned tensor_d0, unsigned tensor_d1,
                                            unsigned stride_elems)
{
  unsigned long long ga = (unsigned long long)(uintptr_t)gaddr;
  u32x4 g0;
  g0[0] = 1u;                                                // count=1 (valid), user mode
  g0[1] = lds_off;                                           // lds_addr (bytes)
  g0[2] = (unsigned)ga;                                      // global_addr[31:0]
  g0[3] = (unsigned)((ga >> 32) & 0x1FFFFFFu) | (2u << 30);  // global_addr[56:32] | type=2
  i32x8 g1;
  g1[0] = (int)((2u << 16) | (1u << 20) | (4u << 22) | (3u << 25));
          // data_size=4B | pad_enable | pad_interval=32 DWORDs | pad_amount=4 DWORDs
  g1[1] = (int)((tensor_d0 & 0xFFFFu) << 16);                           // dim0[15:0]
  g1[2] = (int)((tensor_d0 >> 16) | ((tensor_d1 & 0xFFFFu) << 16));     // dim0[31:16] dim1[15:0]
  g1[3] = (int)((tensor_d1 >> 16) | (tile_d0 << 16));                   // dim1[31:16] tile_d0
  g1[4] = (int)tile_d1;                                                 // tile_d1, tile_d2=0 (2D)
  g1[5] = (int)stride_elems;                                            // dim0_stride[31:0]
  g1[6] = 0;                                                            // stride[47:32], dim1_stride lo
  g1[7] = 0;
  i32x4 gz = {};
#if __clang_major__ >= 23
  i32x8 gz8 = {};
  __builtin_amdgcn_tensor_load_to_lds(g0, g1, gz, gz, gz8, 0);
#else
  __builtin_amdgcn_tensor_load_to_lds(g0, g1, gz, gz, 0);
#endif
}
#else
// Fallback: CDNA5 async global->LDS copy, 16B per lane, tracked by ASYNCcnt.
__device__ __forceinline__ void async_b128(unsigned lds_off, const float* g) {
  asm volatile("global_load_async_to_lds_b128 %0, %1, off"
               :: "v"(lds_off), "v"(g) : "memory");
}
__device__ __forceinline__ void wait_async_all() {
  asm volatile("s_wait_asynccnt 0x0" ::: "memory");
}
__device__ __forceinline__ void wait_async_12() {
  asm volatile("s_wait_asynccnt 0xc" ::: "memory");
}
#endif

// C[M,N] = A[M,K] @ B[N,K]^T (+bias), both A and Bsrc row-major with K contiguous.
__global__ __launch_bounds__(THREADS)
void cp_wmma_gemm(const float* __restrict__ A, const float* __restrict__ Bsrc,
                  const float* __restrict__ bias, float* __restrict__ C,
                  int M, int N, int K, int lda, int ldb)
{
  __shared__ __align__(16) float sA[2][BM][BK + PAD];
  __shared__ __align__(16) float sB[2][BN][BK + PAD];

  const int tid  = threadIdx.x;
  const int wave = tid >> 5;
  const int lane = tid & 31;
  const int lo   = lane & 15;
  const int hi   = lane >> 4;

  const int m0 = blockIdx.y * BM;
  const int n0 = blockIdx.x * BN;

  v8f acc[2][4] = {};

#if USE_TDM
  // Wave 0 drives the Tensor Data Mover; 2 descriptors per K-tile.
  auto issue_tiles = [&](int buf, int k0) {
    if (wave == 0) {
      tdm_load_2d((unsigned)(uintptr_t)&sA[buf][0][0],
                  A + (size_t)m0 * lda + k0, BK, BM,
                  (unsigned)K, (unsigned)M, (unsigned)lda);
      tdm_load_2d((unsigned)(uintptr_t)&sB[buf][0][0],
                  Bsrc + (size_t)n0 * ldb + k0, BK, BN,
                  (unsigned)K, (unsigned)N, (unsigned)ldb);
    }
  };
#else
  const int rb = tid >> 3;   // row slot (0..15)
  const int q  = tid & 7;    // float4 slot within 32-float row
  auto issue_tiles = [&](int buf, int k0) {
    const float* Ag = A + (size_t)(m0 + rb) * lda + k0 + q * 4;
    #pragma unroll
    for (int i = 0; i < 8; ++i) {
      unsigned off = (unsigned)(uintptr_t)&sA[buf][rb + i * 16][q * 4];
      async_b128(off, Ag + (size_t)i * 16 * lda);
    }
    const float* Bg = Bsrc + (size_t)(n0 + rb) * ldb + k0 + q * 4;
    #pragma unroll
    for (int i = 0; i < 4; ++i) {
      unsigned off = (unsigned)(uintptr_t)&sB[buf][rb + i * 16][q * 4];
      async_b128(off, Bg + (size_t)i * 16 * ldb);
    }
  };
#endif

  const int niter = K / BK;
  issue_tiles(0, 0);

  for (int kt = 0; kt < niter; ++kt) {
    const int cur = kt & 1;
    if (kt + 1 < niter) {
      issue_tiles(cur ^ 1, (kt + 1) * BK);   // prefetch next tile into other buffer
#if USE_TDM
      __builtin_amdgcn_s_wait_tensorcnt(2);  // in-order: current tile's 2 DMAs done
#else
      wait_async_12();
#endif
    } else {
#if USE_TDM
      __builtin_amdgcn_s_wait_tensorcnt(0);
#else
      wait_async_all();
#endif
    }
    __syncthreads();

    // B fragments (bf16 32x16): lane = column n, K = hi*16 .. hi*16+15
    v16bf bfrag[4];
    #pragma unroll
    for (int t = 0; t < 4; ++t) {
      const float* p = &sB[cur][t * 16 + lo][hi * 16];
      float4 b0 = *(const float4*)(p);
      float4 b1 = *(const float4*)(p + 4);
      float4 b2 = *(const float4*)(p + 8);
      float4 b3 = *(const float4*)(p + 12);
      pack8(bfrag[t], 0, b0, b1);
      pack8(bfrag[t], 8, b2, b3);
    }

    // A fragments (bf16 16x32): lane = row, K = hi*16 .. hi*16+15
    #pragma unroll
    for (int a = 0; a < 2; ++a) {
      int row = wave * 32 + a * 16 + lo;
      const float* p0 = &sA[cur][row][hi * 16];
      const float* p1 = &sA[cur][row][hi * 16 + 8];
      float4 a0 = *(const float4*)(p0);
      float4 a1 = *(const float4*)(p0 + 4);
      float4 a2 = *(const float4*)(p1);
      float4 a3 = *(const float4*)(p1 + 4);
      v16bf afrag;
      pack8(afrag, 0, a0, a1);
      pack8(afrag, 8, a2, a3);
      #pragma unroll
      for (int t = 0; t < 4; ++t) {
        acc[a][t] = __builtin_amdgcn_wmma_f32_16x16x32_bf16(
            false, afrag, false, bfrag[t], (short)0, acc[a][t], false, false);
      }
    }
    __syncthreads();  // all waves done reading before buffers are overwritten
  }

  // Epilogue: C layout VGPR r -> M = hi*8 + r, N = lo
  float bv[4];
  #pragma unroll
  for (int t = 0; t < 4; ++t)
    bv[t] = bias ? bias[n0 + t * 16 + lo] : 0.0f;

  #pragma unroll
  for (int a = 0; a < 2; ++a) {
    #pragma unroll
    for (int t = 0; t < 4; ++t) {
      int col = n0 + t * 16 + lo;
      #pragma unroll
      for (int r = 0; r < 8; ++r) {
        int row = m0 + wave * 32 + a * 16 + hi * 8 + r;
        C[(size_t)row * N + col] = acc[a][t][r] + bv[t];
      }
    }
  }
}

// U1t[r][k] = U1[k][r] * lam[r]  (32x32 LDS tiles, coalesced both ways)
__global__ __launch_bounds__(256)
void transpose_scale(const float* __restrict__ U1, const float* __restrict__ lam,
                     float* __restrict__ U1t, int IN, int R)
{
  __shared__ float tile[32][33];
  const int tx = threadIdx.x & 31;
  const int ty = threadIdx.x >> 5;  // 0..7
  const int k0 = blockIdx.x * 32;
  const int r0 = blockIdx.y * 32;

  #pragma unroll
  for (int i = 0; i < 4; ++i)
    tile[ty + i * 8][tx] = U1[(size_t)(k0 + ty + i * 8) * R + r0 + tx];
  __syncthreads();
  #pragma unroll
  for (int i = 0; i < 4; ++i) {
    int r = r0 + ty + i * 8;
    U1t[(size_t)r * IN + k0 + tx] = tile[tx][ty + i * 8] * lam[r];
  }
}

extern "C" void kernel_launch(void* const* d_in, const int* in_sizes, int n_in,
                              void* d_out, int out_size, void* d_ws, size_t ws_size,
                              hipStream_t stream) {
  const float* x    = (const float*)d_in[0];  // [B, IN]
  const float* U1   = (const float*)d_in[1];  // [IN, R]
  const float* U2   = (const float*)d_in[2];  // [OUT, R]
  const float* lam  = (const float*)d_in[3];  // [R]
  const float* bias = (const float*)d_in[4];  // [OUT]
  float* out = (float*)d_out;

  const int R   = in_sizes[3];
  const int OUT = in_sizes[4];
  const int IN  = in_sizes[1] / R;
  const int B   = in_sizes[0] / IN;

  float* T   = (float*)d_ws;                  // [B, R]
  float* U1t = T + (size_t)B * R;             // [R, IN]

  // Stage 0: U1t = (U1 * lam)^T
  transpose_scale<<<dim3(IN / 32, R / 32), 256, 0, stream>>>(U1, lam, U1t, IN, R);

  // Stage 1: T = x @ U1t^T   (A: [B x IN], B: [R x IN])
  cp_wmma_gemm<<<dim3(R / BN, B / BM), THREADS, 0, stream>>>(
      x, U1t, nullptr, T, B, R, IN, IN, IN);

  // Stage 2: out = T @ U2^T + bias   (A: [B x R], B: [OUT x R])
  cp_wmma_gemm<<<dim3(OUT / BN, B / BM), THREADS, 0, stream>>>(
      T, U2, bias, out, B, OUT, R, R, R);
}
